// LSTM_35442070126726
// MI455X (gfx1250) — matrix operand order repaired and checked by reading, not compiled, not run on hardware
//
#include <hip/hip_runtime.h>
#include <hip/hip_bf16.h>

// ---------------------------------------------------------------------------
// LSTM  B=32 Q=20 T=100 D=512 H=512    (wave32, gfx1250 WMMA bf16 + async LDS)
// ---------------------------------------------------------------------------
#define TT 100
#define NN 640          // Q*B sequences, n = b*Q + q
#define DD 512
#define HH 512
#define G4 2048         // 4*H, gate order i,f,g,o

typedef __bf16  bf16x4   __attribute__((ext_vector_type(4)));
typedef __bf16  bf16x8   __attribute__((ext_vector_type(8)));
typedef __bf16  bf16x16  __attribute__((ext_vector_type(16)));
typedef float   f32x4    __attribute__((ext_vector_type(4)));
typedef float   f32x8    __attribute__((ext_vector_type(8)));
typedef unsigned int u32x4 __attribute__((ext_vector_type(4)));

// 8 contiguous bf16 (16 bytes) -> one b128 load (global or ds)
__device__ __forceinline__ bf16x8 load_bf8(const __bf16* p) {
    return __builtin_bit_cast(bf16x8, *(const u32x4*)p);
}
__device__ __forceinline__ bf16x16 concat16(bf16x8 lo, bf16x8 hi) {
    return __builtin_shufflevector(lo, hi, 0,1,2,3,4,5,6,7,8,9,10,11,12,13,14,15);
}
__device__ __forceinline__ float sigm(float x) {
    return __builtin_amdgcn_rcpf(1.0f + __expf(-x));
}
__device__ __forceinline__ float tanh_f(float x) {
    return 1.0f - 2.0f * __builtin_amdgcn_rcpf(__expf(2.0f * x) + 1.0f);
}

// ---------------------------------------------------------------------------
// Kernel 1: convert weights f32 -> bf16, fold biases.  1M threads.
// ---------------------------------------------------------------------------
__global__ void lstm_convert(const float* __restrict__ Wih_f,
                             const float* __restrict__ Whh_f,
                             const float* __restrict__ bih,
                             const float* __restrict__ bhh,
                             __bf16* __restrict__ Wih,
                             __bf16* __restrict__ Whh,
                             float* __restrict__ bias) {
    int i = blockIdx.x * blockDim.x + threadIdx.x;   // 0 .. 2048*512-1
    Wih[i] = (__bf16)Wih_f[i];
    Whh[i] = (__bf16)Whh_f[i];
    if (i < G4) bias[i] = bih[i] + bhh[i];
}

// ---------------------------------------------------------------------------
// Kernel 2: x_gates[t*NN+n][j] = sum_d x[n][t][d]*W_ih[j][d] + bias[j]
// One 1024-thread block (32 waves) per 16-row tile; A staged in LDS (bf16),
// each wave owns one 64-wide column group; B tiles preloaded 4-deep so the
// WMMA burst overlaps the next tile's loads.
// ---------------------------------------------------------------------------
__global__ void __launch_bounds__(1024) lstm_xgemm(
        const float*  __restrict__ x,     // [NN][TT][DD] f32
        const __bf16* __restrict__ Wih,   // [G4][DD] bf16
        const float*  __restrict__ bias,  // [G4]
        float*        __restrict__ xg) {  // [TT*NN][G4] f32
    __shared__ alignas(16) __bf16 abuf[16 * DD];   // 16 KB A tile (bf16)

    const int tid  = threadIdx.x;
    const int lane = tid & 31;
    const int half = lane >> 4;      // 0/1 -> K sub-block
    const int mr   = lane & 15;      // row (A) / col (B,C)
    const int cg   = tid >> 5;       // wave = 64-col group 0..31
    const int row0 = blockIdx.x * 16;   // rows share t since NN%16==0
    const int t    = row0 / NN;
    const int n0   = row0 - t * NN;
    const int kbA  = half * 8;       // A lane K base (lanes16-31 start at K=8)
    const int kbB  = half * 8;       // B lane K base (lanes16-31 start at K=8)

    // --- stage A tile: 16 rows x 512 f32, coalesced float4 + cvt -> LDS bf16
    #pragma unroll
    for (int c = tid; c < 2048; c += 1024) {        // 2048 float4 chunks
        const int row = c >> 7;                     // 128 chunks per row
        const int c4  = (c & 127) * 4;
        f32x4 v = *(const f32x4*)(x + ((size_t)(n0 + row) * TT + t) * DD + c4);
        bf16x4 pk;
        pk[0] = (__bf16)v[0]; pk[1] = (__bf16)v[1];
        pk[2] = (__bf16)v[2]; pk[3] = (__bf16)v[3];
        *(bf16x4*)(abuf + row * DD + c4) = pk;      // ds_store_b64
    }
    __syncthreads();

    f32x8 acc[4];
    #pragma unroll
    for (int ct = 0; ct < 4; ++ct) {
        float bv = bias[cg * 64 + ct * 16 + mr];    // depends on n only
        #pragma unroll
        for (int v = 0; v < 8; ++v) acc[ct][v] = bv;
    }

    for (int kc = 0; kc < DD; kc += 32) {
        // A 16x32 bf16 fragment from LDS: two ds_load_b128
        const __bf16* ap = abuf + mr * DD + kc + kbA;
        bf16x16 a = concat16(load_bf8(ap), load_bf8(ap + 16));
        // preload all 4 B tiles so WMMAs overlap remaining loads
        bf16x16 b[4];
        #pragma unroll
        for (int ct = 0; ct < 4; ++ct) {
            const __bf16* bp = Wih + (size_t)(cg * 64 + ct * 16 + mr) * DD + kc + kbB;
            b[ct] = concat16(load_bf8(bp), load_bf8(bp + 16));
        }
        #pragma unroll
        for (int ct = 0; ct < 4; ++ct)
            acc[ct] = __builtin_amdgcn_wmma_f32_16x16x32_bf16(
                false, a, false, b[ct], (short)0, acc[ct], false, false);
    }

    #pragma unroll
    for (int ct = 0; ct < 4; ++ct) {
        const int col = cg * 64 + ct * 16 + mr;
        #pragma unroll
        for (int v = 0; v < 8; ++v) {
            const int m = v + half * 8;
            xg[(size_t)(row0 + m) * G4 + col] = acc[ct][v];
        }
    }
}

// ---------------------------------------------------------------------------
// Kernel 3: 100-step recurrence. 40 WGs x 16 rows; 16 waves each own a
// 32-wide H slice across all 4 gates (8 tiles) so i/f/g/o combine in-register.
// x_gates tiles (128 KB/step) are double-buffered in LDS via
// global_load_async_to_lds_b128 issued one timestep ahead (ASYNCcnt).
// ---------------------------------------------------------------------------
#define XBUF_BYTES (16 * G4 * 4)     // 128 KB per buffer

// Issue 16 async b128 copies: rows j=0..15 of the tile, lane covers 16 bytes.
__device__ __forceinline__ void stage_xg_async(const float* gbase,
                                               unsigned ldsbase, int tid) {
    const unsigned voff = (unsigned)tid * 16u;     // byte offset within a row
    #pragma unroll
    for (int j = 0; j < 16; ++j) {
        const unsigned long long sbase =
            (unsigned long long)(uintptr_t)(gbase + (size_t)j * G4);
        const unsigned laddr = ldsbase + (unsigned)j * (G4 * 4) + voff;
        asm volatile("global_load_async_to_lds_b128 %0, %1, %2"
                     :: "v"(laddr), "v"(voff), "s"(sbase)
                     : "memory");
    }
}

__global__ void __launch_bounds__(512) lstm_rec(
        const float*  __restrict__ xg,    // [TT*NN][G4] f32
        const __bf16* __restrict__ Whh,   // [G4][HH] bf16
        float*        __restrict__ out) { // [NN][HH] f32
    extern __shared__ float xdyn[];                  // 2 x 16 x 2048 f32
    __shared__ alignas(16) __bf16 hbuf[16 * HH];     // 16 KB

    const int tid  = threadIdx.x;
    const int w    = tid >> 5;       // 0..15 wave
    const int lane = tid & 31;
    const int half = lane >> 4;
    const int mr   = lane & 15;
    const int rowb = blockIdx.x * 16;   // sequences rowb..rowb+15
    const int hcb  = w * 32;            // this wave's H slice
    const int kbA  = half * 8;
    const int kbB  = half * 8;
    const unsigned xlds = (unsigned)(uintptr_t)xdyn; // low 32 bits = LDS offset

    for (int i = tid; i < 16 * HH; i += 512) hbuf[i] = (__bf16)0.0f;

    float cre[2][8];
    #pragma unroll
    for (int tt = 0; tt < 2; ++tt)
        #pragma unroll
        for (int v = 0; v < 8; ++v) cre[tt][v] = 0.0f;

    // prefetch t=0 tile into buffer 0
    stage_xg_async(xg + (size_t)rowb * G4, xlds, tid);
    __syncthreads();

    for (int t = 0; t < TT; ++t) {
        // issue next step's tile, then wait for current tile (in-order returns)
        if (t + 1 < TT) {
            stage_xg_async(xg + ((size_t)(t + 1) * NN + rowb) * G4,
                           xlds + ((t + 1) & 1) * XBUF_BYTES, tid);
            asm volatile("s_wait_asynccnt 0x10" ::: "memory");
        } else {
            asm volatile("s_wait_asynccnt 0x0" ::: "memory");
        }
        __syncthreads();   // all waves' staged data visible

        // --- init accumulators from staged x-gates (bias already folded) ---
        const float* xb = xdyn + (size_t)(t & 1) * (16 * G4);
        f32x8 acc[4][2];
        #pragma unroll
        for (int gi = 0; gi < 4; ++gi)
            #pragma unroll
            for (int tt = 0; tt < 2; ++tt) {
                const int col = gi * HH + hcb + tt * 16 + mr;
                #pragma unroll
                for (int v = 0; v < 8; ++v) {
                    const int m = v + half * 8;
                    acc[gi][tt][v] = xb[m * G4 + col];   // ds_load_b32
                }
            }

        // --- gates += h @ W_hh^T  (K = 512), B preloaded 8-deep per k-step ---
        for (int kc = 0; kc < HH; kc += 32) {
            const __bf16* ap = hbuf + mr * HH + kc + kbA;
            bf16x16 a = concat16(load_bf8(ap), load_bf8(ap + 16));
            bf16x16 b[8];
            #pragma unroll
            for (int i = 0; i < 8; ++i) {
                const int gi = i >> 1, tt = i & 1;
                const __bf16* bp = Whh +
                    (size_t)(gi * HH + hcb + tt * 16 + mr) * HH + kc + kbB;
                b[i] = concat16(load_bf8(bp), load_bf8(bp + 16));
            }
            #pragma unroll
            for (int i = 0; i < 8; ++i) {
                const int gi = i >> 1, tt = i & 1;
                acc[gi][tt] = __builtin_amdgcn_wmma_f32_16x16x32_bf16(
                    false, a, false, b[i], (short)0, acc[gi][tt], false, false);
            }
        }
        __syncthreads();   // everyone done reading h[t-1]

        // --- elementwise LSTM cell update (all 4 gates are local) ---
        #pragma unroll
        for (int tt = 0; tt < 2; ++tt)
            #pragma unroll
            for (int v = 0; v < 8; ++v) {
                const int m  = v + half * 8;
                const int hc = hcb + tt * 16 + mr;
                const float iv = acc[0][tt][v];
                const float fv = acc[1][tt][v];
                const float gv = acc[2][tt][v];
                const float ov = acc[3][tt][v];
                const float c  = sigm(fv) * cre[tt][v] + sigm(iv) * tanh_f(gv);
                const float h  = sigm(ov) * tanh_f(c);
                cre[tt][v] = c;
                hbuf[m * HH + hc] = (__bf16)h;
                if (t == TT - 1) out[(size_t)(rowb + m) * HH + hc] = h;
            }
        __syncthreads();   // h[t] visible before next step's reads
    }
}

// ---------------------------------------------------------------------------
extern "C" void kernel_launch(void* const* d_in, const int* in_sizes, int n_in,
                              void* d_out, int out_size, void* d_ws, size_t ws_size,
                              hipStream_t stream) {
    const float* x     = (const float*)d_in[0];  // [B][Q][T][D]
    const float* Wih_f = (const float*)d_in[1];  // [4H][D]
    const float* Whh_f = (const float*)d_in[2];  // [4H][H]
    const float* bih   = (const float*)d_in[3];
    const float* bhh   = (const float*)d_in[4];
    float* out = (float*)d_out;

    char* ws = (char*)d_ws;
    __bf16* Wih  = (__bf16*)(ws);                      // 2 MB
    __bf16* Whh  = (__bf16*)(ws + (size_t)2097152);    // 2 MB
    float*  bias = (float*) (ws + (size_t)4194304);    // 8 KB
    float*  xg   = (float*) (ws + (size_t)4194304 + 8192); // 500 MB [T*N][4H]

    lstm_convert<<<(G4 * DD) / 256, 256, 0, stream>>>(Wih_f, Whh_f, bih, bhh,
                                                      Wih, Whh, bias);
    // one block per 16-row tile (4000), 32 waves = 32 col groups
    lstm_xgemm<<<4000, 1024, 0, stream>>>(x, Wih, bias, xg);
    // 40 persistent WGs, double-buffered async x_gates staging (256 KB dynamic)
    lstm_rec<<<NN / 16, 512, 2 * XBUF_BYTES, stream>>>(xg, Whh, out);
}
